// RNN_10007273799923
// MI455X (gfx1250) — matrix operand order repaired and checked by reading, not compiled, not run on hardware
//
#include <hip/hip_runtime.h>
#include <hip/hip_bf16.h>

typedef __attribute__((ext_vector_type(16))) __bf16 v16bf;
typedef __attribute__((ext_vector_type(8)))  __bf16 v8bf;
typedef __attribute__((ext_vector_type(8)))  float  v8f;

#define LSTM_T 512
#define LSTM_B 64
#define LSTM_D 1024
#define LSTM_H 1024
#define LSTM_G 4096   // 4*H

// CDNA5 async global->LDS path (ASYNCcnt-tracked, no VGPR bounce). Guarded so the
// file still compiles (sync fallback) if this toolchain doesn't declare them.
#if __has_builtin(__builtin_amdgcn_global_load_async_to_lds_b128) && \
    __has_builtin(__builtin_amdgcn_s_wait_asynccnt)
#define LSTM_ASYNC 1
#else
#define LSTM_ASYNC 0
#endif

__device__ __forceinline__ float sigf(float x) {
    return 1.0f / (1.0f + __expf(-x));
}

// stage one 32-byte chunk (16 bf16) global -> LDS
__device__ __forceinline__ void stage32B(const __bf16* g, __bf16* l) {
#if LSTM_ASYNC
    typedef int v4i_ __attribute__((vector_size(16)));
    typedef __attribute__((address_space(1))) v4i_* gp_t;   // global (prints as __device__)
    typedef __attribute__((address_space(3))) v4i_* lp_t;   // LDS
    __builtin_amdgcn_global_load_async_to_lds_b128((gp_t)g, (lp_t)l, 0, 0);
    __builtin_amdgcn_global_load_async_to_lds_b128((gp_t)(g + 8), (lp_t)(l + 8), 0, 0);
#else
    *(v16bf*)l = *(const v16bf*)g;
#endif
}

__device__ __forceinline__ void wait_async_le(int pending) {
#if LSTM_ASYNC
    if (pending) __builtin_amdgcn_s_wait_asynccnt(4);
    else         __builtin_amdgcn_s_wait_asynccnt(0);
#else
    (void)pending;
#endif
}

// ---------------- elementwise helpers ----------------
__global__ void cvt_f32_to_bf16(const float* __restrict__ src, __bf16* __restrict__ dst, int n) {
    int i = blockIdx.x * blockDim.x + threadIdx.x;
    if (i < n) dst[i] = (__bf16)src[i];
}

__global__ void init_state(__bf16* __restrict__ h0, float* __restrict__ c, int n) {
    int i = blockIdx.x * blockDim.x + threadIdx.x;
    if (i < n) { h0[i] = (__bf16)0.0f; c[i] = 0.0f; }
}

// ---------------- xg = x @ W_ih^T + (b_ih + b_hh) ----------------
// A = x_bf16 [M=T*B, K=1024] row-major, B = W_ih_bf16 [N=4096, K=1024] row-major (NT GEMM).
// Block tile 128(M) x 128(N); K staged 32-wide through DOUBLE-BUFFERED LDS with async
// global->LDS loads: chunk k+1 is in flight while WMMAs consume chunk k.
__global__ __launch_bounds__(256) void xg_gemm(
        const __bf16* __restrict__ X, const __bf16* __restrict__ W,
        const float* __restrict__ b_ih, const float* __restrict__ b_hh,
        __bf16* __restrict__ XG) {
    __shared__ __align__(128) __bf16 As[2][128 * 32];
    __shared__ __align__(128) __bf16 Bs[2][128 * 32];

    const int tid  = threadIdx.x;
    const int lane = tid & 31;
    const int wave = tid >> 5;                 // 0..7 -> M tile within block
    const int m0   = blockIdx.y * 128;
    const int n0   = blockIdx.x * 128;
    const int colL = lane & 15;                // A row within tile / B,D column within tile
    const int kLo  = (lane < 16) ? 0 : 16;     // A-matrix K sub-chunk select
    const int kHi  = (lane < 16) ? 0 : 16;     // B-matrix K half select
    const int vOff = (lane < 16) ? 0 : 8;      // C/D row offset

    const int lrow  = tid >> 1;                // 0..127 cooperative stage row
    const int lhalf = (tid & 1) * 16;          // 0 or 16 (bf16 elements)

    const __bf16* gA = &X[(size_t)(m0 + lrow) * LSTM_D + lhalf];
    const __bf16* gB = &W[(size_t)(n0 + lrow) * LSTM_D + lhalf];
    __bf16* lA0 = &As[0][lrow * 32 + lhalf];
    __bf16* lB0 = &Bs[0][lrow * 32 + lhalf];
    const int lbufStride = 128 * 32;

    v8f acc[8] = {};

    // prologue: stage chunk 0 into buffer 0
    stage32B(gA, lA0);
    stage32B(gB, lB0);

    for (int k0 = 0, kk = 0; k0 < LSTM_D; k0 += 32, ++kk) {
        const int buf = kk & 1;
        const int pending = (k0 + 32 < LSTM_D);
        if (pending) {
            // issue next chunk into the other buffer (4 async b128 per thread)
            const int nxt = (buf ^ 1) * lbufStride;
            stage32B(gA + k0 + 32, lA0 + nxt);
            stage32B(gB + k0 + 32, lB0 + nxt);
        }
        wait_async_le(pending);   // current chunk complete (<=4 keeps next in flight)
        __syncthreads();          // chunk `buf` visible to all waves

        // A fragment: lane = row colL of this wave's M tile; K = {kLo..kLo+15}
        const __bf16* arow = &As[buf][(wave * 16 + colL) * 32];
        v8bf alo = *(const v8bf*)(arow + kLo);
        v8bf ahi = *(const v8bf*)(arow + 8 + kLo);
        v16bf a;
#pragma unroll
        for (int i = 0; i < 8; ++i) { a[i] = alo[i]; a[i + 8] = ahi[i]; }

#pragma unroll
        for (int n = 0; n < 8; ++n) {
            v16bf b = *(const v16bf*)&Bs[buf][(n * 16 + colL) * 32 + kHi];
            acc[n] = __builtin_amdgcn_wmma_f32_16x16x32_bf16(
                false, a, false, b, (short)0, acc[n], false, false);
        }
        __syncthreads();          // all waves done reading `buf` before it is restaged
    }

    // epilogue: + bias, store bf16
#pragma unroll
    for (int n = 0; n < 8; ++n) {
        const int col  = n0 + n * 16 + colL;
        const float bv = b_ih[col] + b_hh[col];
#pragma unroll
        for (int v = 0; v < 8; ++v) {
            const int m = m0 + wave * 16 + vOff + v;
            XG[(size_t)m * LSTM_G + col] = (__bf16)(acc[n][v] + bv);
        }
    }
}

// ---------------- one LSTM timestep ----------------
// gates[64,4096] = xg[t] + h_in @ W_hh^T ; then i,f,g,o activations, c/h update.
// 64 blocks (16 h-columns each) x 4 waves (16 rows each). Each wave: 4 gate accumulators.
// xg contribution is added in the epilogue so its loads do not gate the first WMMA.
__global__ __launch_bounds__(128) void lstm_step(
        const __bf16* __restrict__ XG, const __bf16* __restrict__ Whh,
        const __bf16* __restrict__ h_in, __bf16* __restrict__ h_out,
        float* __restrict__ cst, float* __restrict__ out, int t) {
    const int lane = threadIdx.x & 31;
    const int wave = threadIdx.x >> 5;     // M tile: rows 16*wave..16*wave+15
    const int c0   = blockIdx.x * 16;      // h-column tile base
    const int colL = lane & 15;
    const int kLo  = (lane < 16) ? 0 : 16;
    const int kHi  = (lane < 16) ? 0 : 16;
    const int vOff = (lane < 16) ? 0 : 8;
    const int m0   = wave * 16;

    // issue xg loads now; they complete under the K loop (gate order i,f,g,o along 4H)
    float xgv[4][8];
#pragma unroll
    for (int g = 0; g < 4; ++g)
#pragma unroll
        for (int v = 0; v < 8; ++v) {
            const int m = m0 + vOff + v;
            xgv[g][v] = (float)XG[((size_t)(t * LSTM_B + m)) * LSTM_G + g * LSTM_H + c0 + colL];
        }

    v8f acc[4] = {};
    const __bf16* arow = h_in + (size_t)(m0 + colL) * LSTM_H;
    const __bf16* brow = Whh + (size_t)(c0 + colL) * LSTM_H + kHi;   // gate g adds g*H*H
#pragma unroll 2
    for (int k0 = 0; k0 < LSTM_H; k0 += 32) {
        v8bf alo = *(const v8bf*)(arow + k0 + kLo);
        v8bf ahi = *(const v8bf*)(arow + k0 + 8 + kLo);
        v16bf a;
#pragma unroll
        for (int i = 0; i < 8; ++i) { a[i] = alo[i]; a[i + 8] = ahi[i]; }
#pragma unroll
        for (int g = 0; g < 4; ++g) {
            v16bf b = *(const v16bf*)(brow + (size_t)g * LSTM_H * LSTM_H + k0);
            acc[g] = __builtin_amdgcn_wmma_f32_16x16x32_bf16(
                false, a, false, b, (short)0, acc[g], false, false);
        }
    }

    // activations + state update (c columns are block-owned -> in-place is safe)
#pragma unroll
    for (int v = 0; v < 8; ++v) {
        const int m   = m0 + vOff + v;
        const int col = c0 + colL;
        const float iv = sigf(acc[0][v] + xgv[0][v]);
        const float fv = sigf(acc[1][v] + xgv[1][v]);
        const float gv = tanhf(acc[2][v] + xgv[2][v]);
        const float ov = sigf(acc[3][v] + xgv[3][v]);
        const size_t sidx = (size_t)m * LSTM_H + col;
        const float cNew = fv * cst[sidx] + iv * gv;
        cst[sidx] = cNew;
        const float hv = ov * tanhf(cNew);
        h_out[sidx] = (__bf16)hv;
        out[((size_t)t * LSTM_B + m) * LSTM_H + col] = hv;
    }
}

extern "C" void kernel_launch(void* const* d_in, const int* in_sizes, int n_in,
                              void* d_out, int out_size, void* d_ws, size_t ws_size,
                              hipStream_t stream) {
    (void)in_sizes; (void)n_in; (void)out_size; (void)ws_size;
    const float* x    = (const float*)d_in[0];
    const float* W_ih = (const float*)d_in[1];
    const float* W_hh = (const float*)d_in[2];
    const float* b_ih = (const float*)d_in[3];
    const float* b_hh = (const float*)d_in[4];
    float* out = (float*)d_out;

    char* ws = (char*)d_ws;
    size_t off = 0;
    auto wsAlloc = [&](size_t bytes) -> void* {
        void* p = ws + off;
        off = (off + bytes + 255) & ~(size_t)255;
        return p;
    };
    __bf16* xbf   = (__bf16*)wsAlloc((size_t)LSTM_T * LSTM_B * LSTM_D * 2);
    __bf16* wihbf = (__bf16*)wsAlloc((size_t)LSTM_G * LSTM_D * 2);
    __bf16* whhbf = (__bf16*)wsAlloc((size_t)LSTM_G * LSTM_H * 2);
    __bf16* xg    = (__bf16*)wsAlloc((size_t)LSTM_T * LSTM_B * LSTM_G * 2);
    __bf16* hbuf0 = (__bf16*)wsAlloc((size_t)LSTM_B * LSTM_H * 2);
    __bf16* hbuf1 = (__bf16*)wsAlloc((size_t)LSTM_B * LSTM_H * 2);
    float*  cst   = (float*)wsAlloc((size_t)LSTM_B * LSTM_H * 4);

    { int n = LSTM_T * LSTM_B * LSTM_D;
      cvt_f32_to_bf16<<<n / 256, 256, 0, stream>>>(x, xbf, n); }
    { int n = LSTM_G * LSTM_D;
      cvt_f32_to_bf16<<<n / 256, 256, 0, stream>>>(W_ih, wihbf, n); }
    { int n = LSTM_G * LSTM_H;
      cvt_f32_to_bf16<<<n / 256, 256, 0, stream>>>(W_hh, whhbf, n); }
    { int n = LSTM_B * LSTM_H;
      init_state<<<n / 256, 256, 0, stream>>>(hbuf0, cst, n); }

    dim3 gg(LSTM_G / 128, (LSTM_T * LSTM_B) / 128);
    xg_gemm<<<gg, 256, 0, stream>>>(xbf, wihbf, b_ih, b_hh, xg);

    for (int t = 0; t < LSTM_T; ++t) {
        __bf16* hin  = (t & 1) ? hbuf1 : hbuf0;
        __bf16* hout = (t & 1) ? hbuf0 : hbuf1;
        lstm_step<<<LSTM_H / 16, 128, 0, stream>>>(xg, whhbf, hin, hout, cst, out, t);
    }
}
